// Model_29875792510993
// MI455X (gfx1250) — matrix operand order repaired and checked by reading, not compiled, not run on hardware
//
#include <hip/hip_runtime.h>

typedef __attribute__((ext_vector_type(16))) _Float16 v16h;
typedef __attribute__((ext_vector_type(8)))  _Float16 v8h;
typedef __attribute__((ext_vector_type(8)))  float    v8f;

#define T_STEPS  2048
#define BATCH    512
#define HD       50
#define BTILE    16          // batch rows per workgroup (1 WMMA M-tile)
#define NBLK     (BATCH / BTILE)    // 32 workgroups
#define NTHREADS 128                // 4 wave32's, one per SIMD
#define A0STR    72          // layer0 A row stride (halfs, K=64 padded +8)
#define ASTR     136         // layer1/2 A row stride (halfs, K=128 padded +8)
#define CSTR     52

// LDS element counts
#define WF0_N    (32 * 512)  // layer0: 2 ksteps * 16 ntiles, 512 halfs/frag
#define WF1_N    (64 * 512)  // layer1/2: 4 ksteps * 16 ntiles
#define WF2_N    (64 * 512)
#define A0_N     (BTILE * A0STR)
#define A1_N     (BTILE * ASTR)
#define A2_N     (BTILE * ASTR)
#define TH_N     (BTILE * CSTR)
#define WL_N     64
#define SMEM_HALFS  (WF0_N + WF1_N + WF2_N + A0_N + A1_N + A2_N)
#define SMEM_FLOATS (TH_N + WL_N)
#define SMEM_BYTES  (SMEM_HALFS * 2 + SMEM_FLOATS * 4)   // ~175 KB < 320 KB/WGP

// Branch-free fast transcendentals: raw v_exp_f32 / v_rcp_f32.
__device__ __forceinline__ float fexp2(float x) { return __builtin_amdgcn_exp2f(x); }
__device__ __forceinline__ float frcpf(float x) { return __builtin_amdgcn_rcpf(x); }
#define LOG2E 1.4426950408889634f
__device__ __forceinline__ float fsig(float x)  { return frcpf(1.0f + fexp2(-LOG2E * x)); }
__device__ __forceinline__ float ftanh(float x) { return 1.0f - 2.0f * frcpf(1.0f + fexp2(2.0f * LOG2E * x)); }

// Pack weights into WMMA B-fragment layout in LDS (one-time).
// B tile (32x16 f16): lane l holds column N=l%16, 16 K values in table order (32B/lane).
// Logical B[k][n], n = q*64 + j (gate q padded to 64 cols, j<50 valid).
__device__ void pack_frags(_Float16* WF, const float* Wih, const float* Whh,
                           int nfrags, bool l0, int tid) {
    for (int idx = tid; idx < nfrags * 512; idx += NTHREADS) {
        int e    = idx & 511;
        int lane = e >> 4, hcol = e & 15;
        int frag = idx >> 9;
        int kt   = frag >> 4, nt = frag & 15;
        int k = kt * 32 + (lane >> 4) * 8 + (hcol & 7) + (hcol & 8) * 2, n = nt * 16 + (lane & 15);
        int q = n >> 6, j = n & 63;
        float v = 0.0f;
        if (j < HD) {
            int gc = q * HD + j;                  // row of [4H, *] weight
            if (l0) {
                if (k == 0)          v = Wih[gc];
                else if (k <= HD)    v = Whh[gc * HD + (k - 1)];
            } else {
                if (k < HD)          v = Wih[gc * HD + k];
                else if (k < 2 * HD) v = Whh[gc * HD + (k - HD)];
            }
        }
        WF[idx] = (_Float16)v;
    }
}

// One fused LSTM layer step. Wave `ng` owns columns j = ng*16 + (lane&15) of
// ALL four gates (N-tiles 4q+ng): after the WMMAs, acc[0..3][v] hold i/f/g/o
// for the same (batch row, column) so the cell update runs in registers and
// c-state never leaves VGPRs. B fragments are staged from LDS in two
// gate-halves (peak 64 live VGPRs) right before an uninterrupted WMMA chain;
// the internal barrier (with its implied s_wait_dscnt 0) fences "all waves
// done reading A" against the h-writes into A. Lanes with jc>=50 compute
// h==0 exactly (zero weights + zero bias + zero c), so their stores are kept
// unconditional but redirected (v_cndmask, no EXEC branch) into pad columns
// never read by the WMMA K-window.
template <int KS>
__device__ __forceinline__ void layer_step(
    const _Float16* __restrict__ A, int aStride,
    const _Float16* __restrict__ WF,
    const float (&bias)[4], float (&creg)[8],
    _Float16* Aself, int selfOff,
    _Float16* Anext, int nextOff,
    float* TH, int ng, int jc, bool jv, int lane) {
    const int p  = lane >> 4;
    const int lm = lane & 15;

    // A fragments (ISA 16-bit A layout: lane group p reads K-chunks
    // [ks*32+p*8 .. +7] and [+16 .. +23]).
    v16h afr[KS];
    const _Float16* arow = A + lm * aStride;
#pragma unroll
    for (int ks = 0; ks < KS; ++ks) {
        const _Float16* ap = arow + ks * 32 + p * 8;
        v8h lo = *(const v8h*)(ap);
        v8h hi = *(const v8h*)(ap + 16);
        afr[ks] = __builtin_shufflevector(lo, hi, 0, 1, 2, 3, 4, 5, 6, 7,
                                          8, 9, 10, 11, 12, 13, 14, 15);
    }
    __syncthreads();

    v8f acc[4];
#pragma unroll
    for (int half = 0; half < 2; ++half) {
        v16h bf[KS][2];
#pragma unroll
        for (int ks = 0; ks < KS; ++ks)
#pragma unroll
            for (int q2 = 0; q2 < 2; ++q2) {
                int nt = (half * 2 + q2) * 4 + ng;
                bf[ks][q2] = *(const v16h*)(WF + ((ks * 16 + nt) << 9) + (lane << 4));
            }
#pragma unroll
        for (int q2 = 0; q2 < 2; ++q2) {
            v8f a = {};
#pragma unroll
            for (int ks = 0; ks < KS; ++ks)
                a = __builtin_amdgcn_wmma_f32_16x16x32_f16(
                    false, afr[ks], false, bf[ks][q2], (short)0, a, false, false);
            acc[half * 2 + q2] = a;
        }
    }

    // Branch-free store-column redirect for pad lanes (pad tail never read).
    const int colSelf = jv ? (selfOff + jc) : (aStride - 8);
    const int colNext = jv ? (nextOff + jc) : (ASTR - 8);
    const int colTH   = jv ? jc : HD;

    // In-register LSTM cell update; C layout: VGPR v -> row p*8 + v.
    const int r0 = p * 8;
#pragma unroll
    for (int v = 0; v < 8; ++v) {
        float vi = acc[0][v] + bias[0];
        float vf = acc[1][v] + bias[1];
        float vg = acc[2][v] + bias[2];
        float vo = acc[3][v] + bias[3];
        float cc = fsig(vf) * creg[v] + fsig(vi) * ftanh(vg);
        creg[v] = cc;
        float h = fsig(vo) * ftanh(cc);
        int b = r0 + v;
        Aself[b * aStride + colSelf] = (_Float16)h;
        if (Anext) Anext[b * ASTR + colNext] = (_Float16)h;
        if (TH)    TH[b * CSTR + colTH] = ftanh(h);
    }
}

__global__ __launch_bounds__(NTHREADS)
void lstm3_fused(const float* __restrict__ x,
                 const float* __restrict__ Wih0, const float* __restrict__ Whh0,
                 const float* __restrict__ b0,
                 const float* __restrict__ Wih1, const float* __restrict__ Whh1,
                 const float* __restrict__ b1,
                 const float* __restrict__ Wih2, const float* __restrict__ Whh2,
                 const float* __restrict__ b2,
                 const float* __restrict__ Wlin,
                 float* __restrict__ out) {
    extern __shared__ char smem[];
    _Float16* WF0 = (_Float16*)smem;
    _Float16* WF1 = WF0 + WF0_N;
    _Float16* WF2 = WF1 + WF1_N;
    _Float16* A0  = WF2 + WF2_N;     // [x | h0 | 0-pad], K=64
    _Float16* A1  = A0 + A0_N;       // [h0 | h1 | 0-pad], K=128
    _Float16* A2  = A1 + A1_N;       // [h1 | h2 | 0-pad], K=128
    float* TH = (float*)(A2 + A2_N); // tanh(h2) for the head
    float* WL = TH + TH_N;

    const int tid  = threadIdx.x;
    const int wave = tid >> 5, lane = tid & 31;
    const int ng   = wave;                   // gate-column slice owner
    const int gb0  = blockIdx.x * BTILE;
    const int jc   = ng * 16 + (lane & 15);  // hidden column within each gate
    const bool jv  = jc < HD;
    const int row  = lane & 15;              // wave-0 feed/head row (dup lanes)

    // ---- one-time setup ----
    pack_frags(WF0, Wih0, Whh0, 32, true,  tid);
    pack_frags(WF1, Wih1, Whh1, 64, false, tid);
    pack_frags(WF2, Wih2, Whh2, 64, false, tid);
    for (int i = tid; i < A0_N; i += NTHREADS) A0[i] = (_Float16)0.0f;
    for (int i = tid; i < A1_N; i += NTHREADS) A1[i] = (_Float16)0.0f;
    for (int i = tid; i < A2_N; i += NTHREADS) A2[i] = (_Float16)0.0f;
    for (int i = tid; i < TH_N; i += NTHREADS) TH[i] = 0.0f;
    if (tid < WL_N) WL[tid] = (tid < HD) ? Wlin[tid] : 0.0f;

    // Per-lane gate biases in registers.
    float bias0[4], bias1[4], bias2[4];
#pragma unroll
    for (int q = 0; q < 4; ++q) {
        bias0[q] = jv ? b0[q * HD + jc] : 0.0f;
        bias1[q] = jv ? b1[q * HD + jc] : 0.0f;
        bias2[q] = jv ? b2[q * HD + jc] : 0.0f;
    }

    // Cell state lives in registers for the whole sequence.
    float c0[8] = {}, c1[8] = {}, c2[8] = {};

    __syncthreads();

    // ---- persistent recurrence over T, all 3 layers + head fused ----
    for (int t = 0; t < T_STEPS; ++t) {
        if (wave == 0) {   // uniform branch; lanes 16-31 duplicate 0-15
            A0[row * A0STR + 0] = (_Float16)x[(size_t)t * BATCH + gb0 + row];
            if (t + 1 < T_STEPS)
                __builtin_prefetch(x + (size_t)(t + 1) * BATCH + gb0 + row, 0, 0);
        }
        __syncthreads();

        layer_step<2>(A0, A0STR, WF0, bias0, c0,
                      A0, 1, A1, 0, nullptr, ng, jc, jv, lane);
        __syncthreads();
        layer_step<4>(A1, ASTR, WF1, bias1, c1,
                      A1, HD, A2, 0, nullptr, ng, jc, jv, lane);
        __syncthreads();
        layer_step<4>(A2, ASTR, WF2, bias2, c2,
                      A2, HD, nullptr, 0, TH, ng, jc, jv, lane);
        __syncthreads();

        // head: out[t,b] = tanh(h2[b,:]) . Wlin  (lanes 16-31 duplicate)
        if (wave == 0) {
            float s = 0.0f;
#pragma unroll
            for (int j = 0; j < HD; ++j) s += TH[row * CSTR + j] * WL[j];
            out[(size_t)t * BATCH + gb0 + row] = s;
        }
        __syncthreads();
    }
}

extern "C" void kernel_launch(void* const* d_in, const int* in_sizes, int n_in,
                              void* d_out, int out_size, void* d_ws, size_t ws_size,
                              hipStream_t stream) {
    (void)in_sizes; (void)n_in; (void)d_ws; (void)ws_size; (void)out_size;
    const float* x    = (const float*)d_in[0];
    const float* Wih0 = (const float*)d_in[1];
    const float* Whh0 = (const float*)d_in[2];
    const float* b0   = (const float*)d_in[3];
    const float* Wih1 = (const float*)d_in[4];
    const float* Whh1 = (const float*)d_in[5];
    const float* b1   = (const float*)d_in[6];
    const float* Wih2 = (const float*)d_in[7];
    const float* Whh2 = (const float*)d_in[8];
    const float* b2   = (const float*)d_in[9];
    const float* Wlin = (const float*)d_in[10];
    float* out = (float*)d_out;

    // Allow >64KB dynamic LDS (WGP supports up to 320KB on gfx1250).
    (void)hipFuncSetAttribute((const void*)lstm3_fused,
                              hipFuncAttributeMaxDynamicSharedMemorySize,
                              SMEM_BYTES);

    lstm3_fused<<<NBLK, NTHREADS, SMEM_BYTES, stream>>>(
        x, Wih0, Whh0, b0, Wih1, Whh1, b1, Wih2, Whh2, b2, Wlin, out);
}
